// MultiHeadAttentionBlock_28003186769992
// MI455X (gfx1250) — matrix operand order repaired and checked by reading, not compiled, not run on hardware
//
#include <hip/hip_runtime.h>
#include <hip/hip_bf16.h>

typedef __bf16 bf16;
typedef __attribute__((ext_vector_type(16))) __bf16 v16bf;
typedef __attribute__((ext_vector_type(8)))  __bf16 v8bf;
typedef __attribute__((ext_vector_type(8)))  float  v8f;
typedef __attribute__((ext_vector_type(4)))  unsigned int u32x4;
typedef __attribute__((ext_vector_type(8)))  int i32x8;
typedef __attribute__((ext_vector_type(4)))  int i32x4;

#define NB   8      // batch
#define NC   512    // channels
#define NN   1024   // spatial (H*W)
#define NH   8      // heads
#define HD   64     // head dim
#define NG   32     // groups

// Tensor Data Mover availability (false in host pass, true for gfx1250 device pass)
#if defined(__has_builtin)
#if __has_builtin(__builtin_amdgcn_tensor_load_to_lds)
#define HAVE_TDM 1
#endif
#endif
#ifndef HAVE_TDM
#define HAVE_TDM 0
#endif

// ---------------------------------------------------------------- WMMA helpers

__device__ __forceinline__ v8f wmma_bf16f32(v16bf a, v16bf b, v8f c) {
  // D = A(16x32 bf16) * B(32x16 bf16) + C(16x16 f32)
  return __builtin_amdgcn_wmma_f32_16x16x32_bf16(false, a, false, b,
                                                 (short)0, c, false, false);
}

// A fragment (16x32 bf16, M x K), source row-major with leading dim lda.
// ISA layout: lanes 0-15 row m=lane, e0..7 -> K=8*half+e, e8..15 -> K=16+8*half+e
__device__ __forceinline__ v16bf load_a_frag(const bf16* tile, int lda, int lane) {
  const int half = (lane >> 4) & 1;
  const int m    = lane & 15;
  const bf16* p = tile + m * lda + 8 * half;
  v8bf lo = *(const v8bf*)(p);        // K = 8*half + [0..7]
  v8bf hi = *(const v8bf*)(p + 16);   // K = 16 + 8*half + [0..7]
  return __builtin_shufflevector(lo, hi, 0,1,2,3,4,5,6,7,8,9,10,11,12,13,14,15);
}

// B fragment (32x16 bf16, K x N), source stored N-major: tileT[n*ldt + k].
// Layout: n = lane&15, e0..7 -> K = 8*half+e, e8..15 -> K = 16+8*half+e (matches A).
__device__ __forceinline__ v16bf load_bT_frag(const bf16* tileT, int ldt, int lane) {
  const int half = (lane >> 4) & 1;
  const int n    = lane & 15;
  const bf16* p = tileT + n * ldt + 8 * half;
  v8bf lo = *(const v8bf*)(p);
  v8bf hi = *(const v8bf*)(p + 16);
  return __builtin_shufflevector(lo, hi, 0,1,2,3,4,5,6,7,8,9,10,11,12,13,14,15);
}

// ---------------------------------------------------------------- small kernels

__global__ void cvt_f32_bf16(const float* __restrict__ in, bf16* __restrict__ out, int n) {
  int i = blockIdx.x * 256 + threadIdx.x;
  if (i < n) out[i] = (bf16)in[i];
}

// One block per (b, group): reduce 16 channels x 1024, write normalized bf16.
__global__ __launch_bounds__(256) void gn_kernel(const float* __restrict__ x,
                                                 const float* __restrict__ gamma,
                                                 const float* __restrict__ beta,
                                                 bf16* __restrict__ hn) {
  const int t = threadIdx.x;
  const int b = blockIdx.x >> 5;
  const int g = blockIdx.x & 31;
  const size_t base = ((size_t)(b * NC + g * 16)) * NN;
  const float* xg = x + base;
  float s = 0.f, ss = 0.f;
  for (int i = t; i < 16 * NN; i += 256) {
    float v = xg[i];
    s += v; ss += v * v;
  }
  __shared__ float rs[256], rss[256];
  rs[t] = s; rss[t] = ss;
  __syncthreads();
  for (int o = 128; o > 0; o >>= 1) {
    if (t < o) { rs[t] += rs[t + o]; rss[t] += rss[t + o]; }
    __syncthreads();
  }
  const float mu  = rs[0] * (1.0f / 16384.0f);
  const float var = rss[0] * (1.0f / 16384.0f) - mu * mu;
  const float inv = rsqrtf(var + 1e-5f);
  bf16* hng = hn + base;
  for (int i = t; i < 16 * NN; i += 256) {
    int c = g * 16 + (i >> 10);
    float v = (xg[i] - mu) * inv;
    hng[i] = (bf16)(v * gamma[c] + beta[c]);
  }
}

// ---------------------------------------------------------------- GEMM (QKV)
// Y[b] (O x N, bf16) = W (O x K, bf16) @ X[b] (K x N, bf16) + bias
// Block tile 128(M) x 64(N), 8 waves; double-buffered LDS.
// Staging: 16B global loads along the contiguous dim; transpose on LDS-store side.

__global__ __launch_bounds__(256) void gemm_bias_bf16(const bf16* __restrict__ W,
                                                      const bf16* __restrict__ Xall,
                                                      const float* __restrict__ bias,
                                                      bf16* __restrict__ Yall,
                                                      int O, int K, int N) {
  const int LDT = 40;                    // padded pitch (16B-aligned rows)
  __shared__ bf16 sA[2][128 * 40];       // A tile, row-major [m][k]
  __shared__ bf16 sB[2][64 * 40];        // B tile, N-major   [n][k]
  const int t = threadIdx.x, wave = t >> 5, lane = t & 31;
  const int half = (lane >> 4) & 1, nl = lane & 15;
  const int o0 = blockIdx.y * 128, n0 = blockIdx.x * 64;
  const bf16* X = Xall + (size_t)blockIdx.z * K * N;
  bf16* Y = Yall + (size_t)blockIdx.z * O * N;

  // per-thread staging coordinates (all int math)
  const int ar0 = (t + 0)   >> 2, ac0 = ((t + 0)   & 3) << 3;   // A seg 0
  const int ar1 = (t + 256) >> 2, ac1 = ((t + 256) & 3) << 3;   // A seg 1
  const int bk  = t >> 3,         bn  = (t & 7) << 3;           // B seg

  auto stage = [&](bf16* dA, bf16* dB, int kk) {
    v8bf ra0 = *(const v8bf*)(W + (o0 + ar0) * K + kk + ac0);   // 16B loads
    v8bf ra1 = *(const v8bf*)(W + (o0 + ar1) * K + kk + ac1);
    v8bf rb  = *(const v8bf*)(X + (kk + bk) * N + n0 + bn);
    *(v8bf*)(&dA[ar0 * LDT + ac0]) = ra0;                       // 16B LDS stores
    *(v8bf*)(&dA[ar1 * LDT + ac1]) = ra1;
#pragma unroll
    for (int e = 0; e < 8; ++e) dB[(bn + e) * LDT + bk] = rb[e]; // transpose
  };

  v8f acc[4];
#pragma unroll
  for (int nt = 0; nt < 4; ++nt)
#pragma unroll
    for (int r = 0; r < 8; ++r) acc[nt][r] = 0.0f;

  const int nk = K >> 5;
  stage(sA[0], sB[0], 0);
  __syncthreads();

  for (int ks = 0; ks < nk; ++ks) {
    const int cur = ks & 1;
    if (ks + 1 < nk) stage(sA[cur ^ 1], sB[cur ^ 1], (ks + 1) << 5);
    if (ks + 2 < nk) {                   // keep L2->WGP pipe warm
      int kp2 = (ks + 2) << 5;
      __builtin_prefetch(W + (o0 + (t >> 1)) * K + kp2, 0, 0);
      __builtin_prefetch(X + (kp2 + (t >> 3)) * N + n0, 0, 0);
    }
    v16bf a = load_a_frag(sA[cur] + wave * 16 * LDT, LDT, lane);
#pragma unroll
    for (int nt = 0; nt < 4; ++nt) {
      v16bf b = load_bT_frag(sB[cur] + nt * 16 * LDT, LDT, lane);
      acc[nt] = wmma_bf16f32(a, b, acc[nt]);
    }
    __syncthreads();
  }
#pragma unroll
  for (int nt = 0; nt < 4; ++nt)
#pragma unroll
    for (int r = 0; r < 8; ++r) {
      int o = o0 + wave * 16 + r + 8 * half;
      Y[o * N + n0 + nt * 16 + nl] = (bf16)(acc[nt][r] + bias[o]);
    }
}

// ------------------------------------------------- GEMM (proj) + bias + residual

__global__ __launch_bounds__(256) void gemm_proj_resid(const bf16* __restrict__ W,
                                                       const bf16* __restrict__ Xall,
                                                       const float* __restrict__ bias,
                                                       const float* __restrict__ resid,
                                                       float* __restrict__ Yall,
                                                       int O, int K, int N) {
  const int LDT = 40;
  __shared__ bf16 sA[2][128 * 40];
  __shared__ bf16 sB[2][64 * 40];
  const int t = threadIdx.x, wave = t >> 5, lane = t & 31;
  const int half = (lane >> 4) & 1, nl = lane & 15;
  const int o0 = blockIdx.y * 128, n0 = blockIdx.x * 64;
  const size_t boff = (size_t)blockIdx.z * O * N;
  const bf16* X = Xall + (size_t)blockIdx.z * K * N;

  const int ar0 = (t + 0)   >> 2, ac0 = ((t + 0)   & 3) << 3;
  const int ar1 = (t + 256) >> 2, ac1 = ((t + 256) & 3) << 3;
  const int bk  = t >> 3,         bn  = (t & 7) << 3;

  auto stage = [&](bf16* dA, bf16* dB, int kk) {
    v8bf ra0 = *(const v8bf*)(W + (o0 + ar0) * K + kk + ac0);
    v8bf ra1 = *(const v8bf*)(W + (o0 + ar1) * K + kk + ac1);
    v8bf rb  = *(const v8bf*)(X + (kk + bk) * N + n0 + bn);
    *(v8bf*)(&dA[ar0 * LDT + ac0]) = ra0;
    *(v8bf*)(&dA[ar1 * LDT + ac1]) = ra1;
#pragma unroll
    for (int e = 0; e < 8; ++e) dB[(bn + e) * LDT + bk] = rb[e];
  };

  v8f acc[4];
#pragma unroll
  for (int nt = 0; nt < 4; ++nt)
#pragma unroll
    for (int r = 0; r < 8; ++r) acc[nt][r] = 0.0f;

  const int nk = K >> 5;
  stage(sA[0], sB[0], 0);
  __syncthreads();

  for (int ks = 0; ks < nk; ++ks) {
    const int cur = ks & 1;
    if (ks + 1 < nk) stage(sA[cur ^ 1], sB[cur ^ 1], (ks + 1) << 5);
    if (ks + 2 < nk) {
      int kp2 = (ks + 2) << 5;
      __builtin_prefetch(W + (o0 + (t >> 1)) * K + kp2, 0, 0);
      __builtin_prefetch(X + (kp2 + (t >> 3)) * N + n0, 0, 0);
    }
    v16bf a = load_a_frag(sA[cur] + wave * 16 * LDT, LDT, lane);
#pragma unroll
    for (int nt = 0; nt < 4; ++nt) {
      v16bf b = load_bT_frag(sB[cur] + nt * 16 * LDT, LDT, lane);
      acc[nt] = wmma_bf16f32(a, b, acc[nt]);
    }
    __syncthreads();
  }
#pragma unroll
  for (int nt = 0; nt < 4; ++nt)
#pragma unroll
    for (int r = 0; r < 8; ++r) {
      int o = o0 + wave * 16 + r + 8 * half;
      size_t idx = boff + o * N + n0 + nt * 16 + nl;
      Yall[idx] = acc[nt][r] + bias[o] + resid[idx];
    }
}

// ---------------------------------------------------------------- fused attention
// Grid (N/128, heads, batch); each wave owns 16 query rows; flash-style over
// 64-key chunks. qkv layout: [b][s*512 + h*64 + c][n] bf16.
// V chunk (64 x 64, stored [c][j], pitch 72) is staged by the Tensor Data Mover:
// 2D tile, data_size=2B, pad 16B after every 128B row => 144B LDS pitch.

__global__ __launch_bounds__(256) void attn_fused(const bf16* __restrict__ qkv,
                                                  bf16* __restrict__ ao) {
  const int LDQ = 72;                     // pitch, 144B rows (16B aligned)
  __shared__ bf16 sQ[128 * 72];           // Q^T tile  [i][c]
  __shared__ bf16 sK[64 * 72];            // K^T chunk [j][c]
  __shared__ bf16 sV[64 * 72];            // V chunk   [c][j]
  __shared__ bf16 sP[8 * 16 * 64];        // per-wave probs [i][j]
  const int t = threadIdx.x, wave = t >> 5, lane = t & 31;
  const int half = (lane >> 4) & 1, nl = lane & 15;
  const int b = blockIdx.z, h = blockIdx.y, i0 = blockIdx.x * 128;
  const bf16* qp = qkv + ((size_t)(b * 1536 +        h * HD)) * NN;
  const bf16* kp = qkv + ((size_t)(b * 1536 +  512 + h * HD)) * NN;
  const bf16* vp = qkv + ((size_t)(b * 1536 + 1024 + h * HD)) * NN;

  // stage Q transposed (64c x 128i), 16B loads along i, scatter on LDS side
  {
    v8bf rq[4];
#pragma unroll
    for (int s = 0; s < 4; ++s) {
      int lin = t + s * 256;              // 0..1023 segments of 8
      int c = lin >> 4, iseg = (lin & 15) << 3;
      rq[s] = *(const v8bf*)(qp + c * NN + i0 + iseg);
    }
#pragma unroll
    for (int s = 0; s < 4; ++s) {
      int lin = t + s * 256;
      int c = lin >> 4, iseg = (lin & 15) << 3;
#pragma unroll
      for (int e = 0; e < 8; ++e) sQ[(iseg + e) * LDQ + c] = rq[s][e];
    }
  }

  v8f acc_o[4];
  float rmax[8], rsum[8];
#pragma unroll
  for (int ct = 0; ct < 4; ++ct)
#pragma unroll
    for (int r = 0; r < 8; ++r) acc_o[ct][r] = 0.f;
#pragma unroll
  for (int r = 0; r < 8; ++r) { rmax[r] = -3.0e38f; rsum[r] = 0.f; }

  bf16* sPw        = sP + wave * 16 * 64;
  const bf16* sQw  = sQ + wave * 16 * LDQ;
  const float scale = 0.125f;             // 1/sqrt(64)

  for (int j0 = 0; j0 < NN; j0 += 64) {
    __syncthreads();                      // prior reads of sK/sV complete
#if HAVE_TDM
    if (t < 32) {                         // wave 0 issues the TDM for V
      const unsigned long long ga =
          (unsigned long long)(uintptr_t)vp + (unsigned long long)j0 * 2ull;
      const unsigned int la = (unsigned int)(uintptr_t)&sV[0];
      u32x4 g0 = {
          1u,                                             // count=1, user mode
          la,                                             // lds_addr (bytes)
          (unsigned int)(ga & 0xFFFFFFFFu),               // global_addr[31:0]
          (unsigned int)((ga >> 32) & 0x01FFFFFFu) | (2u << 30)  // [56:32]|type=2
      };
      i32x8 g1 = {
          (int)((1u << 16) | (1u << 20) | (4u << 22) | (3u << 25)),
          // data_size=2B, pad_enable, pad_interval=32 DW (128B), pad_amount=4 DW
          (int)(1024u << 16),   // tensor_dim0 = 1024 (j length), barrier addr 0
          (int)(64u << 16),     // tensor_dim0 hi=0, tensor_dim1 = 64 (c length)
          (int)(64u << 16),     // tensor_dim1 hi=0, tile_dim0 = 64 (j)
          64,                   // tile_dim1 = 64 (c), tile_dim2 = 0
          1024,                 // tensor_dim0_stride = 1024 elements
          0, 0                  // stride hi / dim1 stride unused (2D)
      };
      i32x4 gz = {0, 0, 0, 0};
#if __clang_major__ >= 23
      i32x8 gz8 = {0, 0, 0, 0, 0, 0, 0, 0};
      __builtin_amdgcn_tensor_load_to_lds(g0, g1, gz, gz, gz8, 0);
#else
      __builtin_amdgcn_tensor_load_to_lds(g0, g1, gz, gz, 0);
#endif
    }
    {                                     // K^T chunk: 16B loads, LDS transpose
      v8bf rk[2];
#pragma unroll
      for (int s = 0; s < 2; ++s) {
        int lin = t + s * 256;            // 0..511 segments of 8
        int c = lin >> 3, jseg = (lin & 7) << 3;
        rk[s] = *(const v8bf*)(kp + c * NN + j0 + jseg);
      }
#pragma unroll
      for (int s = 0; s < 2; ++s) {
        int lin = t + s * 256;
        int c = lin >> 3, jseg = (lin & 7) << 3;
#pragma unroll
        for (int e = 0; e < 8; ++e) sK[(jseg + e) * LDQ + c] = rk[s][e];
      }
    }
    if (j0 + 64 < NN) {                   // prefetch next K chunk row
      __builtin_prefetch(kp + (t & 63) * NN + j0 + 64, 0, 0);
    }
    if (t < 32) __builtin_amdgcn_s_wait_tensorcnt(0);
#else
    {                                     // fallback: manual K^T and V staging
      v8bf rk[2], rv[2];
#pragma unroll
      for (int s = 0; s < 2; ++s) {
        int lin = t + s * 256;
        int c = lin >> 3, jseg = (lin & 7) << 3;
        rk[s] = *(const v8bf*)(kp + c * NN + j0 + jseg);
        rv[s] = *(const v8bf*)(vp + c * NN + j0 + jseg);
      }
#pragma unroll
      for (int s = 0; s < 2; ++s) {
        int lin = t + s * 256;
        int c = lin >> 3, jseg = (lin & 7) << 3;
#pragma unroll
        for (int e = 0; e < 8; ++e) sK[(jseg + e) * LDQ + c] = rk[s][e];
        *(v8bf*)(&sV[c * LDQ + jseg]) = rv[s];   // V is contiguous both sides
      }
    }
#endif
    __syncthreads();

    // scores: 16(i) x 64(j), K = 64
    v8f s[4];
#pragma unroll
    for (int jt = 0; jt < 4; ++jt)
#pragma unroll
      for (int r = 0; r < 8; ++r) s[jt][r] = 0.f;
#pragma unroll
    for (int kc = 0; kc < HD; kc += 32) {
      v16bf a = load_a_frag(sQw + kc, LDQ, lane);
#pragma unroll
      for (int jt = 0; jt < 4; ++jt) {
        v16bf bb = load_bT_frag(sK + jt * 16 * LDQ + kc, LDQ, lane);
        s[jt] = wmma_bf16f32(a, bb, s[jt]);
      }
    }

    // online softmax: row (m = r + 8*half) lives across the 16 lanes of a half
    float m8[8];
#pragma unroll
    for (int r = 0; r < 8; ++r) {
      float m = -3.0e38f;
#pragma unroll
      for (int jt = 0; jt < 4; ++jt) { s[jt][r] *= scale; m = fmaxf(m, s[jt][r]); }
      m8[r] = m;
    }
#pragma unroll
    for (int off = 8; off > 0; off >>= 1)
#pragma unroll
      for (int r = 0; r < 8; ++r) m8[r] = fmaxf(m8[r], __shfl_xor(m8[r], off, 32));

    float psum[8];
#pragma unroll
    for (int r = 0; r < 8; ++r) {
      float nm   = fmaxf(rmax[r], m8[r]);
      float corr = __expf(rmax[r] - nm);
      rmax[r] = nm;
      rsum[r] *= corr;
#pragma unroll
      for (int ct = 0; ct < 4; ++ct) acc_o[ct][r] *= corr;
      psum[r] = 0.f;
    }
#pragma unroll
    for (int jt = 0; jt < 4; ++jt)
#pragma unroll
      for (int r = 0; r < 8; ++r) {
        float p = __expf(s[jt][r] - rmax[r]);
        psum[r] += p;
        sPw[(r + 8 * half) * 64 + jt * 16 + nl] = (bf16)p;  // D-layout -> [i][j]
      }
#pragma unroll
    for (int off = 8; off > 0; off >>= 1)
#pragma unroll
      for (int r = 0; r < 8; ++r) psum[r] += __shfl_xor(psum[r], off, 32);
#pragma unroll
    for (int r = 0; r < 8; ++r) rsum[r] += psum[r];

    // wave-private LDS relayout: make probs visible to all lanes of this wave
    asm volatile("s_wait_dscnt 0" ::: "memory");

    // O += P (16 x 64) @ V^T (64 x 64): K-dim = j
#pragma unroll
    for (int js = 0; js < 64; js += 32) {
      v16bf a = load_a_frag(sPw + js, 64, lane);
#pragma unroll
      for (int ct = 0; ct < 4; ++ct) {
        v16bf bb = load_bT_frag(sV + ct * 16 * LDQ + js, LDQ, lane);
        acc_o[ct] = wmma_bf16f32(a, bb, acc_o[ct]);
      }
    }
  }

  // normalize + store out^T as [b][h*64+c][i] bf16 (8 contiguous i per lane)
  float rinv[8];
#pragma unroll
  for (int r = 0; r < 8; ++r) rinv[r] = 1.0f / rsum[r];
  const int iBase = i0 + wave * 16 + 8 * half;
#pragma unroll
  for (int ct = 0; ct < 4; ++ct) {
    v8bf pk;
#pragma unroll
    for (int r = 0; r < 8; ++r) pk[r] = (bf16)(acc_o[ct][r] * rinv[r]);
    int ch = h * HD + ct * 16 + nl;
    *(v8bf*)(ao + ((size_t)(b * NC + ch)) * NN + iBase) = pk;
  }
}

// ---------------------------------------------------------------- launcher

extern "C" void kernel_launch(void* const* d_in, const int* in_sizes, int n_in,
                              void* d_out, int out_size, void* d_ws, size_t ws_size,
                              hipStream_t stream) {
  const float* x      = (const float*)d_in[0];
  const float* gamma  = (const float*)d_in[1];
  const float* beta   = (const float*)d_in[2];
  const float* w_qkv  = (const float*)d_in[3];
  const float* b_qkv  = (const float*)d_in[4];
  const float* w_proj = (const float*)d_in[5];
  const float* b_proj = (const float*)d_in[6];
  float* out = (float*)d_out;

  char* ws = (char*)d_ws;
  bf16* hn   = (bf16*)(ws + 0);          //  8 MB: normalized activations bf16
  bf16* wq   = (bf16*)(ws + 8388608);    //  1.5 MB: w_qkv bf16
  bf16* wp   = (bf16*)(ws + 9961472);    //  0.5 MB: w_proj bf16
  bf16* qkvb = (bf16*)(ws + 10485760);   // 24 MB: qkv bf16
  bf16* ao   = (bf16*)(ws + 35651584);   //  8 MB: attention output bf16

  cvt_f32_bf16<<<(1536 * 512) / 256, 256, 0, stream>>>(w_qkv, wq, 1536 * 512);
  cvt_f32_bf16<<<(512 * 512) / 256, 256, 0, stream>>>(w_proj, wp, 512 * 512);
  gn_kernel<<<NB * NG, 256, 0, stream>>>(x, gamma, beta, hn);
  gemm_bias_bf16<<<dim3(NN / 64, 1536 / 128, NB), 256, 0, stream>>>(
      wq, hn, b_qkv, qkvb, 1536, NC, NN);
  attn_fused<<<dim3(NN / 128, NH, NB), 256, 0, stream>>>(qkvb, ao);
  gemm_proj_resid<<<dim3(NN / 64, NC / 128, NB), 256, 0, stream>>>(
      wp, ao, b_proj, x, out, NC, NC, NN);
}
